// TopKRouter_56684978373120
// MI455X (gfx1250) — matrix operand order repaired and checked by reading, not compiled, not run on hardware
//
#include <hip/hip_runtime.h>
#include <hip/hip_bf16.h>

typedef __attribute__((ext_vector_type(16))) __bf16 v16bf;
typedef __attribute__((ext_vector_type(8)))  __bf16 v8bf;
typedef __attribute__((ext_vector_type(16))) float  v16f;
typedef __attribute__((ext_vector_type(8)))  float  v8f;
typedef __attribute__((ext_vector_type(4)))  unsigned uint32x4;
typedef __attribute__((ext_vector_type(8)))  int      int32x8;
typedef __attribute__((ext_vector_type(4)))  int      int32x4;

#define N_TOKENS_C 32768
#define D_MODEL_C  4096
#define N_EXPERTS_C 64
#define BLOCK_M    128
#define KSTEP      64          // two 16x16x32 K-slabs per iteration
#define KITERS     (D_MODEL_C / KSTEP)

#if __has_builtin(__builtin_amdgcn_tensor_load_to_lds)
#define USE_TDM 1
#else
#define USE_TDM 0
#endif

// bf16 vector with both uint4 (LDS I/O) and v16bf (WMMA operand) views
union CVT {
    v16bf v;
    uint4 q[2];
    unsigned u[8];
};

// ---------------- prep kernel: W f32 [64,4096] -> bf16 [64,4096] in d_ws ----------------
__global__ __launch_bounds__(256)
void convert_w_bf16(const float* __restrict__ W, unsigned* __restrict__ out) {
    int tid = blockIdx.x * 256 + threadIdx.x;     // 32768 threads, 8 floats each
    const float* src = W + (size_t)tid * 8;
    float4 f0 = *(const float4*)(src);
    float4 f1 = *(const float4*)(src + 4);
    v8f ff;
    ff[0]=f0.x; ff[1]=f0.y; ff[2]=f0.z; ff[3]=f0.w;
    ff[4]=f1.x; ff[5]=f1.y; ff[6]=f1.z; ff[7]=f1.w;
    union { v8bf v; uint4 q; } c;
    c.v = __builtin_convertvector(ff, v8bf);      // v_cvt_pk_bf16_f32, RNE
    ((uint4*)out)[tid] = c.q;
}

#if USE_TDM
// Issue one TDM tile load: W_bf16 tile [64 experts x 64 k] -> LDS (rows of 128B).
// D# per CDNA5 ISA ch.8: group0 {count, lds_addr, global_addr, type=2},
// group1 {data_size=2B, tensor_dim0=4096, tensor_dim1=64, tile0=64, tile1=64, dim0_stride=4096}.
__device__ __forceinline__ void tdm_load_w_slice(const void* gptr, unsigned lds_off) {
    unsigned long long gaddr = (unsigned long long)(uintptr_t)gptr;
    unsigned long long g0q0 = 1ull | ((unsigned long long)lds_off << 32);          // count=1 | lds_addr
    unsigned long long g0q1 = (gaddr & 0x01FFFFFFFFFFFFFFull) | (2ull << 62);      // global_addr[56:0] | type=2
    const unsigned long long tdim0 = D_MODEL_C, tdim1 = N_EXPERTS_C;
    const unsigned long long tile0 = KSTEP, tile1 = N_EXPERTS_C;
    const unsigned long long stride0 = D_MODEL_C;
    unsigned long long g1q0 = (1ull << 16)                       // data_size = 1 (2 bytes)
                            | ((tdim0 & 0xFFFFull) << 48);       // tensor_dim0[15:0] @ bits 63:48
    unsigned long long g1q1 = (tdim0 >> 16)                      // tensor_dim0[31:16] @ bits 79:64
                            | (tdim1 << 16)                      // tensor_dim1 @ bits 111:80
                            | (tile0 << 48);                     // tile_dim0 @ bits 127:112
    unsigned long long g1q2 = tile1                              // tile_dim1 @ bits 143:128
                            | ((stride0 & 0xFFFFFFFFull) << 32); // dim0_stride[31:0] @ bits 191:160
    unsigned long long g1q3 = (stride0 >> 32) & 0xFFFFull;       // dim0_stride[47:32] @ bits 207:192
    uint32x4 grp0 = { (unsigned)g0q0, (unsigned)(g0q0 >> 32),
                      (unsigned)g0q1, (unsigned)(g0q1 >> 32) };
    int32x8  grp1 = { (int)g1q0, (int)(g1q0 >> 32), (int)g1q1, (int)(g1q1 >> 32),
                      (int)g1q2, (int)(g1q2 >> 32), (int)g1q3, (int)(g1q3 >> 32) };
    int32x4  z4   = { 0, 0, 0, 0 };
#if defined(__clang_major__) && __clang_major__ >= 23
    int32x8  z8   = { 0, 0, 0, 0, 0, 0, 0, 0 };
    __builtin_amdgcn_tensor_load_to_lds(grp0, grp1, z4, z4, z8, 0);
#else
    __builtin_amdgcn_tensor_load_to_lds(grp0, grp1, z4, z4, 0);
#endif
}
#endif

__global__ __launch_bounds__(256)
void topk_router_kernel(const float* __restrict__ inp,
                        const unsigned* __restrict__ wbf,   // bf16 W [64,4096], packed pairs
                        const float* __restrict__ bias,
                        float* __restrict__ probs,
                        int*   __restrict__ topk_idx) {
    // Double-buffered bf16 W k-slice: 2 x 64 experts x 64 k x 2B = 16 KB
    __shared__ unsigned wbuf[2][N_EXPERTS_C * (KSTEP / 2)];
    __shared__ float    sbias[N_EXPERTS_C];
    __shared__ float    sc[BLOCK_M * 65];     // 65-dword row stride: conflict-free epilogue

    const int t    = threadIdx.x;
    const int lane = t & 31;
    const int wave = t >> 5;        // 8 waves
    const int g    = lane >> 4;     // half-wave group (K phase per ISA A/B layouts)
    const int ln16 = lane & 15;

    // SGPR-provable wave id -> s_cbranch, so waves 1..7 truly skip the TDM issue
    // (TDM ignores EXEC; an exec-predicated guard would make every wave issue it).
    const int wave_s = __builtin_amdgcn_readfirstlane(wave);

    const int m_blk = blockIdx.x * BLOCK_M;
    const int row   = m_blk + wave * 16 + ln16;       // token row this lane feeds into A
    const float* arow = inp + (size_t)row * D_MODEL_C;

    if (t < N_EXPERTS_C) sbias[t] = bias[t];

    v8f acc[4] = {};                                  // 16 tokens x 64 experts, f32

    // ---- stage prologue: slice k=0 into buffer 0 ----
#if USE_TDM
    if (wave_s == 0)
        tdm_load_w_slice((const char*)wbf, (unsigned)(uintptr_t)&wbuf[0][0]);
#else
    {
        const int we = t >> 2;                        // expert
        const unsigned* src = wbf + (size_t)we * (D_MODEL_C / 2) + (t & 3) * 8;
        uint4* dst = (uint4*)&wbuf[0][we * (KSTEP / 2) + (t & 3) * 8];
        dst[0] = ((const uint4*)src)[0];
        dst[1] = ((const uint4*)src)[1];
    }
#endif

    int p = 0;
    for (int it = 0; it < KITERS; ++it, p ^= 1) {
        const int k0 = it * KSTEP;

        // ---- 1) kick off DMA of next slice into p^1 (overlaps with A loads below) ----
#if USE_TDM
        if (wave_s == 0 && it + 1 < KITERS)
            tdm_load_w_slice((const char*)wbf + (size_t)(k0 + KSTEP) * 2,
                             (unsigned)(uintptr_t)&wbuf[p ^ 1][0]);
#else
        if (it + 1 < KITERS) {
            const int we = t >> 2;
            const unsigned* src = wbf + (size_t)we * (D_MODEL_C / 2)
                                      + (k0 + KSTEP) / 2 + (t & 3) * 8;
            uint4* dst = (uint4*)&wbuf[p ^ 1][we * (KSTEP / 2) + (t & 3) * 8];
            dst[0] = ((const uint4*)src)[0];
            dst[1] = ((const uint4*)src)[1];
        }
#endif

        // ---- 2) load + convert BOTH A fragments up front (16-bit A 16x32 layout) ----
        // lanes 0-15: K = kh+{0..7,16..23}; lanes 16-31: K = kh+{8..15,24..31}
        CVT afr[2];
        #pragma unroll
        for (int h = 0; h < 2; ++h) {
            const float* ap = arow + k0 + h * 32 + g * 8;
            v16f af;
            float4 c0 = *(const float4*)(ap);
            float4 c1 = *(const float4*)(ap + 4);
            float4 c2 = *(const float4*)(ap + 16);
            float4 c3 = *(const float4*)(ap + 20);
            af[0]=c0.x;  af[1]=c0.y;  af[2]=c0.z;  af[3]=c0.w;
            af[4]=c1.x;  af[5]=c1.y;  af[6]=c1.z;  af[7]=c1.w;
            af[8]=c2.x;  af[9]=c2.y;  af[10]=c2.z; af[11]=c2.w;
            af[12]=c3.x; af[13]=c3.y; af[14]=c3.z; af[15]=c3.w;
            afr[h].v = __builtin_convertvector(af, v16bf);   // v_cvt_pk_bf16_f32
        }
        if (k0 + KSTEP < D_MODEL_C)
            __builtin_prefetch(arow + k0 + KSTEP + g * 8, 0, 3);  // global_prefetch next A

        // ---- 3) make sure slice p's DMA has landed, then publish ----
#if USE_TDM
        if (wave_s == 0) {
            if (it + 1 < KITERS) __builtin_amdgcn_s_wait_tensorcnt(1);  // p done; p^1 in flight
            else                 __builtin_amdgcn_s_wait_tensorcnt(0);
        }
#endif
        __syncthreads();   // publish slice p

        // ---- 4) two K=32 slabs: B preload (DS clause) + 4 WMMAs back-to-back ----
        const unsigned* wslice = &wbuf[p][0];
        #pragma unroll
        for (int h = 0; h < 2; ++h) {
            CVT b[4];
            #pragma unroll
            for (int j = 0; j < 4; ++j) {
                const unsigned* bp =
                    wslice + (j * 16 + ln16) * (KSTEP / 2) + h * 16 + g * 4;
                b[j].q[0] = *(const uint4*)(bp);
                b[j].q[1] = *(const uint4*)(bp + 8);
            }
            #pragma unroll
            for (int j = 0; j < 4; ++j) {
                acc[j] = __builtin_amdgcn_wmma_f32_16x16x32_bf16(
                    false, afr[h].v, false, b[j].v, (short)0, acc[j], false, false);
            }
        }
        __syncthreads();   // all reads of slice p done; p may be overwritten next iter
    }

    // Scatter accumulators: C/D layout -> (m = wave*16 + r + 8*g, n = j*16 + ln16)
    #pragma unroll
    for (int j = 0; j < 4; ++j) {
        #pragma unroll
        for (int r = 0; r < 8; ++r) {
            int m = wave * 16 + r + 8 * g;
            int n = j * 16 + ln16;
            sc[m * 65 + n] = acc[j][r];
        }
    }
    __syncthreads();

    // Fused bias + top-2 + softmax epilogue: one thread per token
    if (t < BLOCK_M) {
        const float* rowp = &sc[t * 65];
        float v0 = -__builtin_inff(), v1 = -__builtin_inff();
        int   i0 = 0, i1 = 0;
        #pragma unroll 8
        for (int e = 0; e < N_EXPERTS_C; ++e) {
            float s = rowp[e] + sbias[e];
            if (s > v0)      { v1 = v0; i1 = i0; v0 = s; i0 = e; }
            else if (s > v1) { v1 = s;  i1 = e; }
        }
        float e1 = __expf(v1 - v0);
        float p0 = 1.0f / (1.0f + e1);
        float p1 = e1 * p0;
        int gtok = m_blk + t;
        probs[gtok * 2 + 0]    = p0;
        probs[gtok * 2 + 1]    = p1;
        topk_idx[gtok * 2 + 0] = i0;
        topk_idx[gtok * 2 + 1] = i1;
    }
}

extern "C" void kernel_launch(void* const* d_in, const int* in_sizes, int n_in,
                              void* d_out, int out_size, void* d_ws, size_t ws_size,
                              hipStream_t stream) {
    (void)in_sizes; (void)n_in; (void)out_size; (void)ws_size;
    const float* inp  = (const float*)d_in[0];
    const float* W    = (const float*)d_in[1];
    const float* bias = (const float*)d_in[2];
    // d_in[3] = top_k (always 2; hardcoded in kernel)

    float* probs = (float*)d_out;                               // [32768, 2] f32
    int*   idx   = (int*)d_out + (size_t)N_TOKENS_C * 2;        // [32768, 2] i32

    unsigned* wbf = (unsigned*)d_ws;   // 512 KB bf16 W, L2-resident

    // 1) one-time W f32 -> bf16 conversion (same stream -> ordered)
    convert_w_bf16<<<dim3((N_EXPERTS_C * D_MODEL_C / 8) / 256), dim3(256), 0, stream>>>(W, wbf);

    // 2) fused GEMM + top-2 + softmax
    dim3 grid(N_TOKENS_C / BLOCK_M);   // 256 blocks
    dim3 block(256);                   // 8 wave32s
    topk_router_kernel<<<grid, block, 0, stream>>>(inp, wbf, bias, probs, idx);
}
